// MultiHeadAttentionForStudy_1202590843455
// MI455X (gfx1250) — matrix operand order repaired and checked by reading, not compiled, not run on hardware
//
#include <hip/hip_runtime.h>
#include <hip/hip_bf16.h>

// ---------------------------------------------------------------------------
// MHA forward for MI455X (gfx1250), wave32 + v_wmma_f32_16x16x32_f16 pipeline,
// software-pipelined (register double-buffered) GEMM loops.
//   B=2, S=2048, D=1024, H=16, dk=64 (fixed by the reference).
// ---------------------------------------------------------------------------

typedef __attribute__((ext_vector_type(16))) _Float16 v16h;
typedef __attribute__((ext_vector_type(8)))  _Float16 v8h;
typedef __attribute__((ext_vector_type(4)))  _Float16 v4h;
typedef __attribute__((ext_vector_type(8)))  float    v8f;

union V16 { v16h v; v8h h[2]; };

#define NB     2
#define SLEN   2048
#define DMODEL 1024
#define NH     16
#define DK     64

__device__ __forceinline__ v8f wmma32(v16h a, v16h b, v8f c) {
  // D = A(16x32 f16) * B(32x16 f16) + C(16x16 f32)
  return __builtin_amdgcn_wmma_f32_16x16x32_f16(
      /*neg_a=*/false, a, /*neg_b=*/false, b,
      /*c_mod=*/(short)0, c, /*reuse_a=*/false, /*reuse_b=*/false);
}

// A-fragment (16x32 f16): lane holds row M=lane&15;
// element h -> K = (lane>>4)*16 + h  => one contiguous 32B load.
__device__ __forceinline__ v16h load_a_frag(const _Float16* __restrict__ p) {
  return *(const v16h*)(p);
}

// ----------------------------- convert kernels -----------------------------

__global__ void cvt_f32_to_f16_x4(const float4* __restrict__ src,
                                  v4h* __restrict__ dst, int n4) {
  int i = blockIdx.x * blockDim.x + threadIdx.x;
  if (i >= n4) return;
  float4 f = src[i];
  v4h h = { (_Float16)f.x, (_Float16)f.y, (_Float16)f.z, (_Float16)f.w };
  dst[i] = h;
}

// Wt[m][h][k][d] = W_m[h][d][k]   (m = 0:q 1:k 2:v), f16
__global__ void cvt_w_qkv(const float* __restrict__ Wq,
                          const float* __restrict__ Wk,
                          const float* __restrict__ Wv,
                          _Float16* __restrict__ Wt) {
  int i = blockIdx.x * blockDim.x + threadIdx.x;   // 3*H*DK*DMODEL
  if (i >= 3 * NH * DK * DMODEL) return;
  int d = i & (DMODEL - 1);
  int t = i >> 10;
  int k = t & (DK - 1); t >>= 6;
  int h = t & (NH - 1); int m = t >> 4;
  const float* W = (m == 0) ? Wq : (m == 1) ? Wk : Wv;
  Wt[i] = (_Float16)W[((size_t)h * DMODEL + d) * DK + k];
}

// WoT[o][d] = Wo[d][o], f16
__global__ void cvt_wo(const float* __restrict__ Wo, _Float16* __restrict__ WoT) {
  int i = blockIdx.x * blockDim.x + threadIdx.x;   // DMODEL*DMODEL
  if (i >= DMODEL * DMODEL) return;
  int d = i & (DMODEL - 1);
  int o = i >> 10;
  WoT[i] = (_Float16)Wo[(size_t)d * DMODEL + o];
}

// --------------------------- QKV projection GEMM ---------------------------
// One wave: 32 rows x 64 cols of one projected matrix for one (b,h).
// Double-buffered K loop: 8 WMMAs per 6 fragment loads.
// q,k stored row-major [b,h,s,dk]; v stored transposed [b,h,dk,s].

__global__ __launch_bounds__(256) void proj_qkv_kernel(
    const _Float16* __restrict__ Qh,   // [B*S, D]
    const _Float16* __restrict__ Wt,   // [3,H,DK,D]
    _Float16* __restrict__ qp,         // [B,H,S,DK]
    _Float16* __restrict__ kp,         // [B,H,S,DK]
    _Float16* __restrict__ vT) {       // [B,H,DK,S]
  const int lane = threadIdx.x & 31;
  const int wave = threadIdx.x >> 5;
  const int half = lane >> 4;
  const int col  = lane & 15;

  int tile = blockIdx.x * 8 + wave;              // 0 .. 3*B*H*(S/32)-1
  const int m     = tile >> 11;                  // 2048 tiles per matrix
  tile &= 2047;
  const int stile = tile & (SLEN / 32 - 1);
  const int bh    = tile >> 6;                   // b*H + h
  const int s0    = stile * 32;

  const _Float16* arow0 = Qh + ((size_t)((bh >> 4) * SLEN + s0 + col)) * DMODEL + half * 16;
  const _Float16* arow1 = arow0 + (size_t)16 * DMODEL;
  const _Float16* wbase = Wt + ((size_t)(m * NH + (bh & 15)) * DK) * DMODEL + half * 16;

  v8f acc[2][4] = {};

  // prologue: fragments for kc = 0
  v16h a0 = load_a_frag(arow0);
  v16h a1 = load_a_frag(arow1);
  v16h w[4];
#pragma unroll
  for (int j = 0; j < 4; ++j)
    w[j] = *(const v16h*)(wbase + (size_t)(j * 16 + col) * DMODEL);

#pragma unroll 2
  for (int kc = 32; kc < DMODEL; kc += 32) {
    // load next chunk into fresh registers (overlaps with WMMAs below)
    v16h a0n = load_a_frag(arow0 + kc);
    v16h a1n = load_a_frag(arow1 + kc);
    v16h wn[4];
#pragma unroll
    for (int j = 0; j < 4; ++j)
      wn[j] = *(const v16h*)(wbase + (size_t)(j * 16 + col) * DMODEL + kc);
    // compute current chunk
#pragma unroll
    for (int j = 0; j < 4; ++j) acc[0][j] = wmma32(a0, w[j], acc[0][j]);
#pragma unroll
    for (int j = 0; j < 4; ++j) acc[1][j] = wmma32(a1, w[j], acc[1][j]);
    a0 = a0n; a1 = a1n;
#pragma unroll
    for (int j = 0; j < 4; ++j) w[j] = wn[j];
  }
  // epilogue chunk
#pragma unroll
  for (int j = 0; j < 4; ++j) acc[0][j] = wmma32(a0, w[j], acc[0][j]);
#pragma unroll
  for (int j = 0; j < 4; ++j) acc[1][j] = wmma32(a1, w[j], acc[1][j]);

  if (m < 2) {
    _Float16* dst = ((m == 0) ? qp : kp) + ((size_t)bh * SLEN + s0) * DK;
#pragma unroll
    for (int rt = 0; rt < 2; ++rt)
#pragma unroll
      for (int j = 0; j < 4; ++j)
#pragma unroll
        for (int r = 0; r < 8; ++r)
          dst[(size_t)(rt * 16 + r + half * 8) * DK + j * 16 + col] = (_Float16)acc[rt][j][r];
  } else {
    _Float16* dst = vT + ((size_t)bh * DK) * SLEN + s0;
#pragma unroll
    for (int rt = 0; rt < 2; ++rt)
#pragma unroll
      for (int j = 0; j < 4; ++j)
#pragma unroll
        for (int r = 0; r < 8; ++r)
          dst[(size_t)(j * 16 + col) * SLEN + rt * 16 + r + half * 8] = (_Float16)acc[rt][j][r];
  }
}

// ---------------------- flash attention (online softmax) -------------------
// One wave owns a 16-row query tile of one (b,h); streams keys in blocks of 32.
// K-fragments for block tb+32 and V-fragments for block tb are issued before
// the softmax VALU section so their latency hides under exp/shuffle work.

__global__ __launch_bounds__(256) void attn_kernel(
    const _Float16* __restrict__ qp,   // [B,H,S,DK]
    const _Float16* __restrict__ kp,   // [B,H,S,DK]
    const _Float16* __restrict__ vT,   // [B,H,DK,S]
    _Float16* __restrict__ concat) {   // [B,S,H*DK]
  __shared__ __align__(32) _Float16 Plds[8 * 16 * 32];   // per-wave 16x32 P tile
  const int lane = threadIdx.x & 31;
  const int wave = threadIdx.x >> 5;
  const int half = lane >> 4;
  const int col  = lane & 15;

  const int tile  = blockIdx.x * 8 + wave;       // 0 .. B*H*(S/16)-1
  const int stile = tile & (SLEN / 16 - 1);
  const int bh    = tile >> 7;
  const int s0    = stile * 16;

  const _Float16* qbase = qp + ((size_t)bh * SLEN) * DK;
  const _Float16* kbase = kp + ((size_t)bh * SLEN) * DK + (size_t)col * DK + half * 16;
  const _Float16* vbase = vT + ((size_t)bh * DK) * SLEN + (size_t)col * SLEN + half * 16;
  _Float16* pl = &Plds[wave * 512];

  // q A-fragments, kept in registers for the whole key sweep (dk=64 -> 2 chunks)
  v16h qa[2];
  {
    const _Float16* qr = qbase + (size_t)(s0 + col) * DK + half * 16;
    qa[0] = load_a_frag(qr);
    qa[1] = load_a_frag(qr + 32);
  }

  float mrow[8], lrow[8];
  v8f acc[4] = {};
#pragma unroll
  for (int r = 0; r < 8; ++r) { mrow[r] = -3.0e38f; lrow[r] = 0.0f; }

  // prologue: key fragments for block 0.  kb[t2][kc]: B(K=feat, N=key) tile.
  v16h kbc[2][2];
#pragma unroll
  for (int t2 = 0; t2 < 2; ++t2)
#pragma unroll
    for (int kc = 0; kc < 2; ++kc)
      kbc[t2][kc] = *(const v16h*)(kbase + (size_t)(t2 * 16) * DK + kc * 32);

  for (int tb = 0; tb < SLEN; tb += 32) {
    // ---- issue V fragments for this block and K fragments for next block ----
    v16h vb[4];
#pragma unroll
    for (int j = 0; j < 4; ++j)
      vb[j] = *(const v16h*)(vbase + (size_t)(j * 16) * SLEN + tb);
    v16h kbn[2][2];
    const bool has_next = (tb + 32 < SLEN);
    if (has_next) {
#pragma unroll
      for (int t2 = 0; t2 < 2; ++t2)
#pragma unroll
        for (int kc = 0; kc < 2; ++kc)
          kbn[t2][kc] = *(const v16h*)(kbase + (size_t)(tb + 32 + t2 * 16) * DK + kc * 32);
    }
    // ---- scores: two 16x16 tiles over this 32-key block ----
    v8f sc[2];
#pragma unroll
    for (int t2 = 0; t2 < 2; ++t2) {
      v8f c = {};
      c = wmma32(qa[0], kbc[t2][0], c);
      c = wmma32(qa[1], kbc[t2][1], c);
      sc[t2] = c * 0.125f;                       // 1/sqrt(dk)
    }
    // ---- online softmax (row = r + half*8, cols striped over 16 lanes) ----
    asm volatile("" ::: "memory");
#pragma unroll
    for (int r = 0; r < 8; ++r) {
      float rmax = fmaxf(sc[0][r], sc[1][r]);
      rmax = fmaxf(rmax, __shfl_xor(rmax, 1, 32));
      rmax = fmaxf(rmax, __shfl_xor(rmax, 2, 32));
      rmax = fmaxf(rmax, __shfl_xor(rmax, 4, 32));
      rmax = fmaxf(rmax, __shfl_xor(rmax, 8, 32));
      float mnew = fmaxf(mrow[r], rmax);
      float corr = __expf(mrow[r] - mnew);
      float p0 = __expf(sc[0][r] - mnew);
      float p1 = __expf(sc[1][r] - mnew);
      float ps = p0 + p1;
      ps += __shfl_xor(ps, 1, 32);
      ps += __shfl_xor(ps, 2, 32);
      ps += __shfl_xor(ps, 4, 32);
      ps += __shfl_xor(ps, 8, 32);
      lrow[r] = lrow[r] * corr + ps;
      mrow[r] = mnew;
#pragma unroll
      for (int j = 0; j < 4; ++j) acc[j][r] *= corr;
      pl[(r + half * 8) * 32 + col]      = (_Float16)p0;
      pl[(r + half * 8) * 32 + 16 + col] = (_Float16)p1;
    }
    asm volatile("" ::: "memory");   // per-wave DS ops are in-order; fence the compiler
    // ---- re-read P in A-fragment layout, then P @ V ----
    V16 up;
    up.h[0] = *(const v8h*)&pl[col * 32 + half * 16];
    up.h[1] = *(const v8h*)&pl[col * 32 + half * 16 + 8];
#pragma unroll
    for (int j = 0; j < 4; ++j)
      acc[j] = wmma32(up.v, vb[j], acc[j]);
    asm volatile("" ::: "memory");
    if (has_next) {
#pragma unroll
      for (int t2 = 0; t2 < 2; ++t2)
#pragma unroll
        for (int kc = 0; kc < 2; ++kc)
          kbc[t2][kc] = kbn[t2][kc];
    }
  }

  // normalize + store into concat [B,S,H*DK]
  _Float16* outb = concat + ((size_t)(bh >> 4) * SLEN + s0) * (NH * DK) + (bh & 15) * DK;
#pragma unroll
  for (int j = 0; j < 4; ++j)
#pragma unroll
    for (int r = 0; r < 8; ++r) {
      float v = acc[j][r] / lrow[r];
      outb[(size_t)(r + half * 8) * (NH * DK) + j * 16 + col] = (_Float16)v;
    }
}

// --------------------------- output projection GEMM ------------------------
// One wave: 32 rows x 64 cols, double-buffered K loop, f32 output.

__global__ __launch_bounds__(256) void out_proj_kernel(
    const _Float16* __restrict__ concat,  // [B*S, D]
    const _Float16* __restrict__ WoT,     // [D_out, D_in]
    float* __restrict__ out) {            // [B*S, D]
  const int lane = threadIdx.x & 31;
  const int wave = threadIdx.x >> 5;
  const int half = lane >> 4;
  const int col  = lane & 15;

  const int tile = blockIdx.x * 8 + wave;   // 0..2047
  const int ct   = tile & 15;               // 16 column tiles of 64
  const int rt   = tile >> 4;               // 128 row tiles of 32

  const _Float16* arow0 = concat + (size_t)(rt * 32 + col) * DMODEL + half * 16;
  const _Float16* arow1 = arow0 + (size_t)16 * DMODEL;
  const _Float16* wbase = WoT + (size_t)(ct * 64) * DMODEL + half * 16;

  v8f acc[2][4] = {};

  v16h a0 = load_a_frag(arow0);
  v16h a1 = load_a_frag(arow1);
  v16h w[4];
#pragma unroll
  for (int j = 0; j < 4; ++j)
    w[j] = *(const v16h*)(wbase + (size_t)(j * 16 + col) * DMODEL);

#pragma unroll 2
  for (int kc = 32; kc < DMODEL; kc += 32) {
    v16h a0n = load_a_frag(arow0 + kc);
    v16h a1n = load_a_frag(arow1 + kc);
    v16h wn[4];
#pragma unroll
    for (int j = 0; j < 4; ++j)
      wn[j] = *(const v16h*)(wbase + (size_t)(j * 16 + col) * DMODEL + kc);
#pragma unroll
    for (int j = 0; j < 4; ++j) acc[0][j] = wmma32(a0, w[j], acc[0][j]);
#pragma unroll
    for (int j = 0; j < 4; ++j) acc[1][j] = wmma32(a1, w[j], acc[1][j]);
    a0 = a0n; a1 = a1n;
#pragma unroll
    for (int j = 0; j < 4; ++j) w[j] = wn[j];
  }
#pragma unroll
  for (int j = 0; j < 4; ++j) acc[0][j] = wmma32(a0, w[j], acc[0][j]);
#pragma unroll
  for (int j = 0; j < 4; ++j) acc[1][j] = wmma32(a1, w[j], acc[1][j]);

#pragma unroll
  for (int rt2 = 0; rt2 < 2; ++rt2)
#pragma unroll
    for (int j = 0; j < 4; ++j)
#pragma unroll
      for (int r = 0; r < 8; ++r)
        out[(size_t)(rt * 32 + rt2 * 16 + r + half * 8) * DMODEL + ct * 64 + j * 16 + col] =
            acc[rt2][j][r];
}

// --------------------------------- launcher --------------------------------

extern "C" void kernel_launch(void* const* d_in, const int* in_sizes, int n_in,
                              void* d_out, int out_size, void* d_ws, size_t ws_size,
                              hipStream_t stream) {
  (void)in_sizes; (void)n_in; (void)out_size; (void)ws_size;
  const float* Q  = (const float*)d_in[0];
  const float* Wq = (const float*)d_in[1];
  const float* Wk = (const float*)d_in[2];
  const float* Wv = (const float*)d_in[3];
  const float* Wo = (const float*)d_in[4];
  float* out = (float*)d_out;

  char* ws = (char*)d_ws;
  const size_t MB = 1024 * 1024;
  _Float16* Qh    = (_Float16*)(ws);             //  8 MiB  [B*S, D]
  _Float16* Wt    = (_Float16*)(ws + 8  * MB);   //  6 MiB  [3,H,DK,D]
  _Float16* WoT   = (_Float16*)(ws + 14 * MB);   //  2 MiB  [D,D]
  _Float16* qp    = (_Float16*)(ws + 16 * MB);   //  8 MiB  [B,H,S,DK]
  _Float16* kp    = (_Float16*)(ws + 24 * MB);   //  8 MiB  [B,H,S,DK]
  _Float16* vT    = (_Float16*)(ws + 32 * MB);   //  8 MiB  [B,H,DK,S]
  _Float16* cat   = (_Float16*)(ws + 40 * MB);   //  8 MiB  [B,S,D]
  // total 48 MiB workspace

  const int nQ4 = NB * SLEN * DMODEL / 4;            // 1,048,576 float4s
  cvt_f32_to_f16_x4<<<(nQ4 + 255) / 256, 256, 0, stream>>>((const float4*)Q, (v4h*)Qh, nQ4);
  const int nW = 3 * NH * DK * DMODEL;               // 3,145,728
  cvt_w_qkv<<<(nW + 255) / 256, 256, 0, stream>>>(Wq, Wk, Wv, Wt);
  const int nWo = DMODEL * DMODEL;                   // 1,048,576
  cvt_wo<<<(nWo + 255) / 256, 256, 0, stream>>>(Wo, WoT);

  // 3 matrices * B*H*(S/32) = 6144 wave-tiles, 8 waves per block
  proj_qkv_kernel<<<6144 / 8, 256, 0, stream>>>(Qh, Wt, qp, kp, vT);
  // B*H*(S/16) = 4096 wave-tiles
  attn_kernel<<<4096 / 8, 256, 0, stream>>>(qp, kp, vT, cat);
  // (B*S/32)*(D/64) = 2048 wave-tiles
  out_proj_kernel<<<2048 / 8, 256, 0, stream>>>(cat, WoT, out);
}
